// CausalTransformer_19516331393401
// MI455X (gfx1250) — matrix operand order repaired and checked by reading, not compiled, not run on hardware
//
#include <hip/hip_runtime.h>
#include <hip/hip_bf16.h>

// ---------------------------------------------------------------------------
// CDNA5 (gfx1250) fused causal-transformer kernel.
// Key identity: softmax rows sum to 1  =>  attention collapses to
//   out = LN(v + vpe_sum)*g + b + v,  v = x @ Wv^T (per head).
// All matmuls run on v_wmma_f32_16x16x32_bf16 (wave32 WMMA, f32 accum).
// ---------------------------------------------------------------------------

typedef __attribute__((ext_vector_type(16))) __bf16        v16bf;
typedef __attribute__((ext_vector_type(8)))  float         v8f;
typedef __attribute__((ext_vector_type(8)))  unsigned int  v8u;

#define Bn 8
#define Ln 1024
#define En 512
#define Hn 8
#define HDn 64
#define EPSC 1e-5f

// ---- bf16 helpers (RNE) ----------------------------------------------------
static __device__ __forceinline__ unsigned short f2bf(float f) {
    unsigned int x = __float_as_uint(f);
    x += 0x7FFFu + ((x >> 16) & 1u);
    return (unsigned short)(x >> 16);
}
static __device__ __forceinline__ unsigned int pack2bf(float a, float b) {
    return (unsigned int)f2bf(a) | ((unsigned int)f2bf(b) << 16);
}

// ---- WMMA fragment loaders -------------------------------------------------
// A-matrix 16x32 bf16 (ISA 7.12.2): lane m=lane&15; kbase=(lane>=16)?8:0;
// VGPR r holds K = (r&3)*2 + (r>>2)*16 + kbase  (+0,+1 in the two halves).
static __device__ __forceinline__ v16bf lds_afrag(const unsigned short* tile,
                                                  int rowStride, int colOff, int lane) {
    const int m     = lane & 15;
    const int kbase = (lane >> 4) << 3;
    const unsigned short* row = tile + m * rowStride + colOff + kbase;
    v8u u;
#pragma unroll
    for (int r = 0; r < 8; ++r) {
        const int k = ((r & 3) << 1) + ((r >> 2) << 4);
        u[r] = *(const unsigned int*)(row + k);
    }
    return __builtin_bit_cast(v16bf, u);
}

// B-matrix 32x16 bf16: pre-packed in register order; one 32B load per lane.
static __device__ __forceinline__ v16bf g_bfrag(const unsigned int* frag, int lane) {
    const v8u* p = (const v8u*)frag;
    return __builtin_bit_cast(v16bf, p[lane]);
}

static __device__ __forceinline__ v8f wmma_bf16(v16bf a, v16bf b, v8f c) {
    return __builtin_amdgcn_wmma_f32_16x16x32_bf16(false, a, false, b,
                                                   (short)0, c, false, false);
}

// ---------------------------------------------------------------------------
// Kernel 0: vpe_sum[l][d] = sum_j values_pos_enc[l][j][d]   (L=1024, d=64)
// ---------------------------------------------------------------------------
__global__ __launch_bounds__(256) void vpe_kernel(const float* __restrict__ vpe,
                                                  float* __restrict__ outp) {
    __shared__ float red[256];
    const int l  = blockIdx.x;
    const int t  = threadIdx.x;
    const int d  = t & 63;
    const int jg = t >> 6;
    const float* base = vpe + (size_t)l * (Ln * HDn);
    float acc = 0.f;
    for (int j = jg; j < Ln; j += 4) acc += base[j * HDn + d];
    red[t] = acc;
    __syncthreads();
    if (t < 64) outp[l * HDn + t] = red[t] + red[t + 64] + red[t + 128] + red[t + 192];
}

// ---------------------------------------------------------------------------
// Kernel 1: pack f32 weight matrices W[N][K] into bf16 B-fragments.
// Fragment layout: [mat][kstep][ntile][lane(32)][dword(8)], 256 dwords each.
// B element: n = nt*16 + (lane&15); k = ks*32 + (lane>=16?8:0) + (r&3)*2 + (r>>2)*16 (+1).
// ---------------------------------------------------------------------------
__global__ __launch_bounds__(256) void pack_b_kernel(const float* __restrict__ src,
                                                     unsigned int* __restrict__ dst,
                                                     int K, int nTiles, int kSteps,
                                                     int matStride) {
    const int frag = blockIdx.x;
    const int nt   = frag % nTiles;
    const int tmp  = frag / nTiles;
    const int ks   = tmp % kSteps;
    const int mat  = tmp / kSteps;

    const float* W = src + (size_t)mat * matStride;
    unsigned int* o = dst + (size_t)frag * 256;

    const int t    = threadIdx.x;
    const int lane = t >> 3;
    const int r    = t & 7;
    const int n    = nt * 16 + (lane & 15);
    const int k    = ks * 32 + ((lane >> 4) << 3) + ((r & 3) << 1) + ((r >> 2) << 4);
    o[lane * 8 + r] = pack2bf(W[(size_t)n * K + k], W[(size_t)n * K + k + 1]);
}

// ---------------------------------------------------------------------------
// Head LayerNorm over hd=64 (C-fragment rows live in 16-lane halves; reduce
// with shfl_xor 1/2/4/8), then  out += LN(acc+vpe)*g + b + acc.
// ---------------------------------------------------------------------------
static __device__ __forceinline__ void head_ln_add(const v8f (&acc)[4],
                                                   const float (&vp)[4][8],
                                                   const float* __restrict__ g,
                                                   const float* __restrict__ b,
                                                   int nL, float (&hsum)[4][8],
                                                   bool first) {
    float gg[4], bb[4];
#pragma unroll
    for (int e = 0; e < 4; ++e) { gg[e] = g[e * 16 + nL]; bb[e] = b[e * 16 + nL]; }
#pragma unroll
    for (int r = 0; r < 8; ++r) {
        float u[4];
        float s = 0.f, q = 0.f;
#pragma unroll
        for (int e = 0; e < 4; ++e) {
            u[e] = acc[e][r] + vp[e][r];
            s += u[e]; q += u[e] * u[e];
        }
        s += __shfl_xor(s, 1, 32); q += __shfl_xor(q, 1, 32);
        s += __shfl_xor(s, 2, 32); q += __shfl_xor(q, 2, 32);
        s += __shfl_xor(s, 4, 32); q += __shfl_xor(q, 4, 32);
        s += __shfl_xor(s, 8, 32); q += __shfl_xor(q, 8, 32);
        const float mean = s * (1.f / 64.f);
        const float var  = q * (1.f / 64.f) - mean * mean;
        const float rs   = rsqrtf(var + EPSC);
#pragma unroll
        for (int e = 0; e < 4; ++e) {
            const float o = (u[e] - mean) * rs * gg[e] + bb[e] + acc[e][r];
            if (first) hsum[e][r] = o; else hsum[e][r] += o;
        }
    }
}

// 16x512 @ 512x512^T tile GEMM on WMMA: wave owns 4 n-tiles (64 columns).
static __device__ __forceinline__ void gemm_tile(const unsigned short* __restrict__ atile,
                                                 const unsigned int* __restrict__ wp,
                                                 const float* __restrict__ bias,
                                                 int blk, int wv, int lane, int nL,
                                                 v8f (&acc)[4]) {
#pragma unroll
    for (int j = 0; j < 4; ++j) {
        const float bb = bias[(wv * 4 + j) * 16 + nL];
#pragma unroll
        for (int r = 0; r < 8; ++r) acc[j][r] = bb;
    }
#pragma unroll
    for (int ks = 0; ks < 16; ++ks) {
        const v16bf a = lds_afrag(atile, En, ks * 32, lane);
#pragma unroll
        for (int j = 0; j < 4; ++j) {
            const v16bf bw = g_bfrag(wp + (size_t)((blk * 16 + ks) * 32 + wv * 4 + j) * 256, lane);
            acc[j] = wmma_bf16(a, bw, acc[j]);
        }
    }
}

// ---------------------------------------------------------------------------
// Kernel 2: fused block:  h = mha(x) + mha(z);  out = LN(relu(h W1^T+b1) W2^T+b2)
// grid = 3 blocks * 8 batches * 64 row-tiles; 256 threads = 8 waves = 8 heads.
// ---------------------------------------------------------------------------
__global__ __launch_bounds__(256) void fused_block_kernel(
        const float* __restrict__ treat, const float* __restrict__ outc,
        const float* __restrict__ cov,
        const float* __restrict__ ln_g, const float* __restrict__ ln_b,
        const float* __restrict__ ffb1, const float* __restrict__ ffb2,
        const float* __restrict__ flg,  const float* __restrict__ flb,
        const float* __restrict__ vpe,
        const unsigned int* __restrict__ wvp,
        const unsigned int* __restrict__ w1p,
        const unsigned int* __restrict__ w2p,
        float* __restrict__ out) {
    // smA: x-tile + z-tile (bf16), later reused for ff1-tile (bf16)
    __shared__ unsigned short smA[2 * 16 * En];   // 32 KB
    __shared__ unsigned short smH[16 * En];       // 16 KB  (h tile, bf16)
    __shared__ float psumS[8][16];
    __shared__ float psumQ[8][16];

    const int tid   = threadIdx.x;
    const int lane  = tid & 31;
    const int wv    = tid >> 5;           // wave id == head id
    const int nL    = lane & 15;
    const int half8 = (lane >> 4) << 3;   // C-frag row offset (0 or 8)

    const int bid   = blockIdx.x;
    const int blk   = bid >> 9;
    const int rem   = bid & 511;
    const int batch = rem >> 6;
    const int l0    = (rem & 63) << 4;

    const float* xs = (blk == 0) ? treat : (blk == 1 ? outc  : cov);
    const float* zs = (blk == 0) ? cov   : (blk == 1 ? treat : outc);

    // -------- stage x/z tiles (16 x 512) as bf16 in LDS --------
    {
        const int row = tid >> 4;
        const int c0  = (tid & 15) << 5;   // 32 floats per thread
        const float2* xr = (const float2*)(xs + ((size_t)batch * Ln + l0 + row) * En + c0);
        const float2* zr = (const float2*)(zs + ((size_t)batch * Ln + l0 + row) * En + c0);
        unsigned int* dx = (unsigned int*)(smA + row * En + c0);
        unsigned int* dz = (unsigned int*)(smA + 16 * En + row * En + c0);
#pragma unroll
        for (int i = 0; i < 16; ++i) {
            float2 xv = xr[i], zv = zr[i];
            dx[i] = pack2bf(xv.x, xv.y);
            dz[i] = pack2bf(zv.x, zv.y);
        }
    }
    __syncthreads();

    // -------- per-head V projections: v = x_h @ Wv^T  (16x64x64, bf16 WMMA) --------
    v8f accx[4], accz[4];
#pragma unroll
    for (int e = 0; e < 4; ++e)
#pragma unroll
        for (int r = 0; r < 8; ++r) { accx[e][r] = 0.f; accz[e][r] = 0.f; }

    const int colx = wv * HDn;
#pragma unroll
    for (int ks = 0; ks < 2; ++ks) {
        const v16bf ax = lds_afrag(smA,            En, colx + ks * 32, lane);
        const v16bf az = lds_afrag(smA + 16 * En,  En, colx + ks * 32, lane);
#pragma unroll
        for (int e = 0; e < 4; ++e) {
            const v16bf bx = g_bfrag(wvp + (size_t)((((blk * 2 + 0) * Hn + wv) * 2 + ks) * 4 + e) * 256, lane);
            const v16bf bz = g_bfrag(wvp + (size_t)((((blk * 2 + 1) * Hn + wv) * 2 + ks) * 4 + e) * 256, lane);
            accx[e] = wmma_bf16(ax, bx, accx[e]);
            accz[e] = wmma_bf16(az, bz, accz[e]);
        }
    }

    // -------- vpe tile (same for every head), head-LN, residual, h = hx + hz --------
    float vp[4][8];
#pragma unroll
    for (int e = 0; e < 4; ++e)
#pragma unroll
        for (int r = 0; r < 8; ++r)
            vp[e][r] = vpe[(size_t)(l0 + r + half8) * HDn + e * 16 + nL];

    float hsum[4][8];
    head_ln_add(accx, vp, ln_g + ((blk * 2 + 0) * Hn + wv) * HDn,
                          ln_b + ((blk * 2 + 0) * Hn + wv) * HDn, nL, hsum, true);
    head_ln_add(accz, vp, ln_g + ((blk * 2 + 1) * Hn + wv) * HDn,
                          ln_b + ((blk * 2 + 1) * Hn + wv) * HDn, nL, hsum, false);

#pragma unroll
    for (int r = 0; r < 8; ++r)
#pragma unroll
        for (int e = 0; e < 4; ++e)
            smH[(r + half8) * En + wv * HDn + e * 16 + nL] = f2bf(hsum[e][r]);
    __syncthreads();

    // -------- FFN layer 1: ff1 = relu(h @ W1^T + b1) --------
    v8f acc1[4];
    gemm_tile(smH, w1p, ffb1 + blk * En, blk, wv, lane, nL, acc1);

    unsigned short* ff1 = smA;   // reuse x/z staging area
#pragma unroll
    for (int j = 0; j < 4; ++j)
#pragma unroll
        for (int r = 0; r < 8; ++r) {
            float v = acc1[j][r];
            v = v > 0.f ? v : 0.f;
            ff1[(r + half8) * En + (wv * 4 + j) * 16 + nL] = f2bf(v);
        }
    __syncthreads();

    // -------- FFN layer 2: ff2 = ff1 @ W2^T + b2 --------
    v8f acc2[4];
    gemm_tile(ff1, w2p, ffb2 + blk * En, blk, wv, lane, nL, acc2);

    // -------- final LayerNorm over E=512 (cross-wave via LDS partials) --------
#pragma unroll
    for (int r = 0; r < 8; ++r) {
        float s = acc2[0][r] + acc2[1][r] + acc2[2][r] + acc2[3][r];
        float q = acc2[0][r] * acc2[0][r] + acc2[1][r] * acc2[1][r]
                + acc2[2][r] * acc2[2][r] + acc2[3][r] * acc2[3][r];
        s += __shfl_xor(s, 1, 32); q += __shfl_xor(q, 1, 32);
        s += __shfl_xor(s, 2, 32); q += __shfl_xor(q, 2, 32);
        s += __shfl_xor(s, 4, 32); q += __shfl_xor(q, 4, 32);
        s += __shfl_xor(s, 8, 32); q += __shfl_xor(q, 8, 32);
        if (nL == 0) { psumS[wv][r + half8] = s; psumQ[wv][r + half8] = q; }
    }
    __syncthreads();

    float fgv[4], fbv[4];
#pragma unroll
    for (int j = 0; j < 4; ++j) {
        fgv[j] = flg[blk * En + (wv * 4 + j) * 16 + nL];
        fbv[j] = flb[blk * En + (wv * 4 + j) * 16 + nL];
    }
    float* outbase = out + ((size_t)blk * Bn + batch) * Ln * En;
#pragma unroll
    for (int r = 0; r < 8; ++r) {
        const int m = r + half8;
        float s = 0.f, q = 0.f;
#pragma unroll
        for (int w = 0; w < 8; ++w) { s += psumS[w][m]; q += psumQ[w][m]; }
        const float mean = s * (1.f / 512.f);
        const float var  = q * (1.f / 512.f) - mean * mean;
        const float rs   = rsqrtf(var + EPSC);
        float* orow = outbase + (size_t)(l0 + m) * En;
#pragma unroll
        for (int j = 0; j < 4; ++j)
            orow[(wv * 4 + j) * 16 + nL] = (acc2[j][r] - mean) * rs * fgv[j] + fbv[j];
    }
}

// ---------------------------------------------------------------------------
extern "C" void kernel_launch(void* const* d_in, const int* in_sizes, int n_in,
                              void* d_out, int out_size, void* d_ws, size_t ws_size,
                              hipStream_t stream) {
    const float* treat = (const float*)d_in[0];
    const float* outc  = (const float*)d_in[1];
    const float* cov   = (const float*)d_in[2];
    // d_in[3] active_entries, d_in[4] keys_pos_enc: provably unused (softmax rows sum to 1)
    const float* vpe_in = (const float*)d_in[5];
    const float* Wv    = (const float*)d_in[6];
    // d_in[7] Wk, d_in[8] Wq: unused (reference projects from v only... and attn collapses)
    const float* ln_g  = (const float*)d_in[9];
    const float* ln_b  = (const float*)d_in[10];
    const float* ffW1  = (const float*)d_in[11];
    const float* ffb1  = (const float*)d_in[12];
    const float* ffW2  = (const float*)d_in[13];
    const float* ffb2  = (const float*)d_in[14];
    const float* flg   = (const float*)d_in[15];
    const float* flb   = (const float*)d_in[16];
    float* outp        = (float*)d_out;

    // workspace layout (bytes, all 1KB-aligned)
    char* ws = (char*)d_ws;
    float*        d_vpe = (float*)(ws + 0);                         //   256 KB
    unsigned int* d_wvp = (unsigned int*)(ws + 262144);             //   384 KB (48 mats * 8 frags)
    unsigned int* d_w1p = (unsigned int*)(ws + 262144 + 393216);    //  1536 KB (3*512 frags)
    unsigned int* d_w2p = (unsigned int*)(ws + 262144 + 393216 + 1572864);

    vpe_kernel<<<Ln, 256, 0, stream>>>(vpe_in, d_vpe);
    // Wv: 3*2*8 matrices of [64][64]: nTiles=4, kSteps=2
    pack_b_kernel<<<48 * 2 * 4, 256, 0, stream>>>(Wv, d_wvp, 64, 4, 2, 64 * 64);
    // W1/W2: 3 matrices of [512][512]: nTiles=32, kSteps=16
    pack_b_kernel<<<3 * 16 * 32, 256, 0, stream>>>(ffW1, d_w1p, 512, 32, 16, 512 * 512);
    pack_b_kernel<<<3 * 16 * 32, 256, 0, stream>>>(ffW2, d_w2p, 512, 32, 16, 512 * 512);

    fused_block_kernel<<<3 * Bn * (Ln / 16), 256, 0, stream>>>(
        treat, outc, cov, ln_g, ln_b, ffb1, ffb2, flg, flb,
        d_vpe, d_wvp, d_w1p, d_w2p, outp);
}
